// EnhancedDualDomainMambaBlock_32315333935260
// MI455X (gfx1250) — matrix operand order repaired and checked by reading, not compiled, not run on hardware
//
#include <hip/hip_runtime.h>
#include <hip/hip_bf16.h>
#include <math.h>

// Problem constants (from reference)
#define B_   4
#define L_   2048
#define D_   512
#define DI_  1024
#define DS_  16
#define DC_  4
#define DTR_ 32
#define H_   4
#define DH_  128
#define BL_  (B_ * L_)
#define XDBL_ (DTR_ + 2 * DS_)   // 64

typedef __attribute__((ext_vector_type(16))) _Float16 v16h;
typedef __attribute__((ext_vector_type(8)))  _Float16 v8h;
typedef __attribute__((ext_vector_type(8)))  float    v8f;

__device__ __forceinline__ float sigmoidf_(float x) { return 1.0f / (1.0f + expf(-x)); }
__device__ __forceinline__ float siluf_(float x)    { return x * sigmoidf_(x); }
__device__ __forceinline__ float geluf_(float x)    { return 0.5f * x * (1.0f + erff(x * 0.70710678118654752f)); }

// Fragment loaders per CDNA5 ISA 7.12.2 (16-bit, wave32):
//   A 16x32: lane(0-15)=row M; regs 0-3 hold K=half*8+0..7, regs 4-7 K=16+half*8+0..7
//   B 32x16: lane holds col N; regs 0-3 hold K=half*8+0..7, regs 4-7 K=16+half*8+0..7
// K-contiguous layout => each fragment is exactly two 16B global loads.
__device__ __forceinline__ v16h load_frag_a(const _Float16* __restrict__ p, int half) {
  v8h a0 = *(const v8h*)(p + half * 8);
  v8h a1 = *(const v8h*)(p + 16 + half * 8);
  v16h av;
#pragma unroll
  for (int i = 0; i < 8; ++i) { av[i] = a0[i]; av[i + 8] = a1[i]; }
  return av;
}
__device__ __forceinline__ v16h load_frag_b(const _Float16* __restrict__ p, int half) {
  v8h b0 = *(const v8h*)(p + half * 8);
  v8h b1 = *(const v8h*)(p + 16 + half * 8);
  v16h bv;
#pragma unroll
  for (int i = 0; i < 8; ++i) { bv[i] = b0[i]; bv[i + 8] = b1[i]; }
  return bv;
}

// ---------------------------------------------------------------------------
// WMMA GEMM: C[M,N] = A[M,K] @ W[N,K]^T (+bias[N]).  A,W fp16 K-major, C fp32.
// Block = 4 waves = 64x64 output tile; each wave does 4 M-tiles x 1 N-tile.
// B fragment reused 4x in registers; all waves in block share the same A rows
// (L0/L2 reuse). global_prefetch_b8 hints issued ahead of the A/B streams.
// M % 64 == 0, N % 64 == 0, K % 32 == 0 for every call site.
// ---------------------------------------------------------------------------
__global__ __launch_bounds__(128)
void wmma_gemm_nt(const _Float16* __restrict__ A, const _Float16* __restrict__ W,
                  const float* __restrict__ bias, float* __restrict__ C,
                  int M, int N, int K) {
  const int lane = threadIdx.x & 31;
  const int wave = threadIdx.x >> 5;
  const int half = lane >> 4;
  const int r    = lane & 15;

  const int n0 = blockIdx.x * 64 + wave * 16;
  const int m0 = blockIdx.y * 64;
  if (n0 >= N || m0 >= M) return;   // wave-uniform; EXEC stays all-ones

  v8f acc[4] = {};
  const _Float16* ap[4];
#pragma unroll
  for (int mt = 0; mt < 4; ++mt)
    ap[mt] = A + (size_t)(m0 + mt * 16 + r) * K;
  const _Float16* brow = W + (size_t)(n0 + r) * K;

  for (int k0 = 0; k0 < K; k0 += 32) {
    // speculative prefetch a few K-steps ahead (global_prefetch_b8; OOB dropped)
    __builtin_prefetch(brow + k0 + 128, 0, 1);
    __builtin_prefetch(ap[0] + k0 + 128, 0, 1);
    v16h bv = load_frag_b(brow + k0, half);
#pragma unroll
    for (int mt = 0; mt < 4; ++mt) {
      v16h av = load_frag_a(ap[mt] + k0, half);
      acc[mt] = __builtin_amdgcn_wmma_f32_16x16x32_f16(
          false, av, false, bv, (short)0, acc[mt], false, false);
    }
  }

  const float bb = bias ? bias[n0 + r] : 0.0f;
#pragma unroll
  for (int mt = 0; mt < 4; ++mt) {
    float* crow = C + (size_t)(m0 + mt * 16 + half * 8) * N + (n0 + r);
#pragma unroll
    for (int g = 0; g < 8; ++g)
      crow[(size_t)g * N] = acc[mt][g] + bb;
  }
}

// ---------------------------------------------------------------------------
// Elementwise / fused helpers
// ---------------------------------------------------------------------------
__global__ void cvt_f32_f16(const float* __restrict__ in, _Float16* __restrict__ out, int n) {
  int i = blockIdx.x * blockDim.x + threadIdx.x;
  if (i < n) out[i] = (_Float16)in[i];
}

// causal depthwise conv (width 4) + bias + silu; writes f32 (for scan) and f16 (for x_proj GEMM)
__global__ void conv_silu_kernel(const float* __restrict__ uz, const float* __restrict__ cw,
                                 const float* __restrict__ cb,
                                 float* __restrict__ ucf, _Float16* __restrict__ uch) {
  int idx = blockIdx.x * blockDim.x + threadIdx.x;
  if (idx >= BL_ * DI_) return;
  int di = idx & (DI_ - 1);
  int bl = idx >> 10;
  int l  = bl & (L_ - 1);
  int b  = bl >> 11;
  float acc = cb[di];
#pragma unroll
  for (int j = 0; j < DC_; ++j) {
    int lj = l - (DC_ - 1) + j;
    if (lj >= 0)
      acc += uz[(size_t)(b * L_ + lj) * (2 * DI_) + di] * cw[di * DC_ + j];
  }
  float s = siluf_(acc);
  ucf[idx] = s;
  uch[idx] = (_Float16)s;
}

__global__ void slice_dt_kernel(const float* __restrict__ xdbl, _Float16* __restrict__ dth) {
  int idx = blockIdx.x * blockDim.x + threadIdx.x;
  if (idx >= BL_ * DTR_) return;
  int c = idx & (DTR_ - 1);
  int row = idx >> 5;
  dth[idx] = (_Float16)xdbl[(size_t)row * XDBL_ + c];
}

__global__ void softplus_kernel(float* __restrict__ delta, const float* __restrict__ dtb) {
  int idx = blockIdx.x * blockDim.x + threadIdx.x;
  if (idx >= BL_ * DI_) return;
  int di = idx & (DI_ - 1);
  float t = delta[idx] + dtb[di];
  delta[idx] = (t > 20.0f) ? t : log1pf(expf(t));
}

// selective scan: one LANE per (b, di, s) -> 65536 lanes (16x more parallel than
// one-thread-per-channel). Lanes 0-15 / 16-31 of a wave each hold one channel's
// 16 SSM states; y = sum_s h*C reduced with 4 shfl_xor steps inside the wave.
// Epilogue (+u*D, *silu(z), ->f16) fused in lane s==0.
__global__ __launch_bounds__(256)
void scan_kernel(const float* __restrict__ ucf, const float* __restrict__ delta,
                 const float* __restrict__ xdbl, const float* __restrict__ A_log,
                 const float* __restrict__ ssmD, const float* __restrict__ uz,
                 _Float16* __restrict__ ygh) {
  int t = blockIdx.x * blockDim.x + threadIdx.x;
  int s  = t & (DS_ - 1);
  int ch = t >> 4;                 // (b, di)
  if (ch >= B_ * DI_) return;
  int di = ch & (DI_ - 1);
  int b  = ch >> 10;
  const float Av = -expf(A_log[di * DS_ + s]);
  const float Dv = ssmD[di];
  float h = 0.0f;
  for (int l = 0; l < L_; ++l) {
    size_t row = (size_t)b * L_ + l;
    float d = delta[row * DI_ + di];
    float u = ucf[row * DI_ + di];
    float Bv = xdbl[row * XDBL_ + DTR_ + s];
    float Cv = xdbl[row * XDBL_ + DTR_ + DS_ + s];
    h = expf(d * Av) * h + d * u * Bv;
    float y = h * Cv;
    y += __shfl_xor(y, 1, 32);
    y += __shfl_xor(y, 2, 32);
    y += __shfl_xor(y, 4, 32);
    y += __shfl_xor(y, 8, 32);
    if (s == 0) {
      y += u * Dv;
      float z = uz[row * (2 * DI_) + DI_ + di];
      ygh[row * DI_ + di] = (_Float16)(y * siluf_(z));
    }
  }
}

__device__ __forceinline__ float block_sum(float v, float* red) {
  int tid = threadIdx.x;
  red[tid] = v;
  __syncthreads();
  for (int s = blockDim.x >> 1; s > 0; s >>= 1) {
    if (tid < s) red[tid] += red[tid + s];
    __syncthreads();
  }
  float r = red[0];
  __syncthreads();
  return r;
}

__device__ __forceinline__ float block_max(float v, float* red) {
  int tid = threadIdx.x;
  red[tid] = v;
  __syncthreads();
  for (int s = blockDim.x >> 1; s > 0; s >>= 1) {
    if (tid < s) red[tid] = fmaxf(red[tid], red[tid + s]);
    __syncthreads();
  }
  float r = red[0];
  __syncthreads();
  return r;
}

// fused: x1n = LN1(x1)*(1+g[b]);  x2 = LN2(residual + x1n); also emit x2 in f16
__global__ __launch_bounds__(128)
void ln1_gate_ln2_kernel(const float* __restrict__ x1, const float* __restrict__ xres,
                         const float* __restrict__ g,
                         const float* __restrict__ ln1w, const float* __restrict__ ln1b,
                         const float* __restrict__ ln2w, const float* __restrict__ ln2b,
                         float* __restrict__ x2, _Float16* __restrict__ x2h) {
  __shared__ float red[128];
  int row = blockIdx.x;
  int b = row >> 11;       // row / L_
  int tid = threadIdx.x;
  const float* xr = x1 + (size_t)row * D_;
  float v[4];
  float s = 0.0f;
#pragma unroll
  for (int i = 0; i < 4; ++i) { v[i] = xr[tid + i * 128]; s += v[i]; }
  float mean = block_sum(s, red) * (1.0f / D_);
  float var = 0.0f;
#pragma unroll
  for (int i = 0; i < 4; ++i) { float d = v[i] - mean; var += d * d; }
  var = block_sum(var, red) * (1.0f / D_);
  float inv = rsqrtf(var + 1e-5f);
  float r2[4];
  float s2 = 0.0f;
#pragma unroll
  for (int i = 0; i < 4; ++i) {
    int d = tid + i * 128;
    float xn = (v[i] - mean) * inv * ln1w[d] + ln1b[d];
    xn *= (1.0f + g[b * D_ + d]);
    float rr = xres[(size_t)row * D_ + d] + xn;
    r2[i] = rr; s2 += rr;
  }
  float mean2 = block_sum(s2, red) * (1.0f / D_);
  float var2 = 0.0f;
#pragma unroll
  for (int i = 0; i < 4; ++i) { float d = r2[i] - mean2; var2 += d * d; }
  var2 = block_sum(var2, red) * (1.0f / D_);
  float inv2 = rsqrtf(var2 + 1e-5f);
#pragma unroll
  for (int i = 0; i < 4; ++i) {
    int d = tid + i * 128;
    float o = (r2[i] - mean2) * inv2 * ln2w[d] + ln2b[d];
    x2[(size_t)row * D_ + d]  = o;
    x2h[(size_t)row * D_ + d] = (_Float16)o;
  }
}

// x3 = LN3(x2 + attno[b])  (attention output broadcast over L)
__global__ __launch_bounds__(128)
void ln3_kernel(const float* __restrict__ x2, const float* __restrict__ attno,
                const float* __restrict__ w, const float* __restrict__ bp,
                float* __restrict__ x3) {
  __shared__ float red[128];
  int row = blockIdx.x;
  int b = row >> 11;
  int tid = threadIdx.x;
  float v[4];
  float s = 0.0f;
#pragma unroll
  for (int i = 0; i < 4; ++i) {
    int d = tid + i * 128;
    v[i] = x2[(size_t)row * D_ + d] + attno[b * D_ + d];
    s += v[i];
  }
  float mean = block_sum(s, red) * (1.0f / D_);
  float var = 0.0f;
#pragma unroll
  for (int i = 0; i < 4; ++i) { float d = v[i] - mean; var += d * d; }
  var = block_sum(var, red) * (1.0f / D_);
  float inv = rsqrtf(var + 1e-5f);
#pragma unroll
  for (int i = 0; i < 4; ++i) {
    int d = tid + i * 128;
    x3[(size_t)row * D_ + d] = (v[i] - mean) * inv * w[d] + bp[d];
  }
}

// tiny dense: Y[R,N] = act(X[R,K] @ W[N,K]^T + bias)   (R == B_ == 4)
__global__ void small_linear_kernel(const float* __restrict__ X, const float* __restrict__ W,
                                    const float* __restrict__ bias, float* __restrict__ Y,
                                    int R, int N, int K, int act) {
  int idx = blockIdx.x * blockDim.x + threadIdx.x;
  if (idx >= R * N) return;
  int n = idx % N, r = idx / N;
  const float* xr = X + (size_t)r * K;
  const float* wr = W + (size_t)n * K;
  float acc = bias ? bias[n] : 0.0f;
  for (int k = 0; k < K; ++k) acc += xr[k] * wr[k];
  if (act == 1) acc = geluf_(acc);
  else if (act == 2) acc = sigmoidf_(acc);
  Y[idx] = acc;
}

// cross-attention with q_len = 1: one block per (b, head)
__global__ __launch_bounds__(256)
void attn_kernel(const float* __restrict__ q, const float* __restrict__ k,
                 const float* __restrict__ v, float* __restrict__ o) {
  __shared__ float sc[L_];
  __shared__ float red[256];
  int b = blockIdx.x / H_;
  int h = blockIdx.x % H_;
  int tid = threadIdx.x;
  const float* qh = q + b * D_ + h * DH_;
  float mx = -1e30f;
  for (int l = tid; l < L_; l += 256) {
    const float* kh = k + (size_t)(b * L_ + l) * D_ + h * DH_;
    float s = 0.0f;
#pragma unroll 4
    for (int d = 0; d < DH_; ++d) s += qh[d] * kh[d];
    s *= 0.08838834764831845f;   // 1/sqrt(DH)
    sc[l] = s;
    mx = fmaxf(mx, s);
  }
  mx = block_max(mx, red);
  float ssum = 0.0f;
  for (int l = tid; l < L_; l += 256) {
    float p = expf(sc[l] - mx);
    sc[l] = p;
    ssum += p;
  }
  ssum = block_sum(ssum, red);
  __syncthreads();
  if (tid < DH_) {
    float acc = 0.0f;
    for (int l = 0; l < L_; ++l)
      acc += sc[l] * v[(size_t)(b * L_ + l) * D_ + h * DH_ + tid];
    o[b * D_ + h * DH_ + tid] = acc / ssum;
  }
}

// grouped (4) temporal conv k=3 pad=1 + BN(eval) + GELU + residual -> out
__global__ void teconv_kernel(const float* __restrict__ x3, const float* __restrict__ w,
                              const float* __restrict__ cb, const float* __restrict__ gamma,
                              const float* __restrict__ beta, const float* __restrict__ mean,
                              const float* __restrict__ var, float* __restrict__ out) {
  int idx = blockIdx.x * blockDim.x + threadIdx.x;
  if (idx >= BL_ * D_) return;
  int c  = idx & (D_ - 1);
  int bl = idx >> 9;
  int l  = bl & (L_ - 1);
  int b  = bl >> 11;
  int ci0 = (c >> 7) << 7;            // group base channel
  float acc = cb[c];
  const float* wc = w + (size_t)c * (D_ / 4) * 3;
#pragma unroll
  for (int t = 0; t < 3; ++t) {
    int lt = l - 1 + t;
    if (lt < 0 || lt >= L_) continue;
    const float* xr = x3 + (size_t)(b * L_ + lt) * D_ + ci0;
    for (int ci = 0; ci < D_ / 4; ++ci)
      acc += xr[ci] * wc[ci * 3 + t];
  }
  float xn = (acc - mean[c]) * rsqrtf(var[c] + 1e-5f) * gamma[c] + beta[c];
  out[idx] = x3[idx] + geluf_(xn);
}

// ---------------------------------------------------------------------------
extern "C" void kernel_launch(void* const* d_in, const int* in_sizes, int n_in,
                              void* d_out, int out_size, void* d_ws, size_t ws_size,
                              hipStream_t stream) {
  (void)in_sizes; (void)n_in; (void)out_size; (void)ws_size;
  const float* x         = (const float*)d_in[0];
  const float* freq      = (const float*)d_in[1];
  const float* in_proj_w = (const float*)d_in[2];
  const float* conv_w    = (const float*)d_in[3];
  const float* conv_b    = (const float*)d_in[4];
  const float* x_proj_w  = (const float*)d_in[5];
  const float* dt_proj_w = (const float*)d_in[6];
  const float* dt_proj_b = (const float*)d_in[7];
  const float* A_log     = (const float*)d_in[8];
  const float* ssm_D     = (const float*)d_in[9];
  const float* out_proj_w= (const float*)d_in[10];
  const float* fg_w1     = (const float*)d_in[11];
  const float* fg_b1     = (const float*)d_in[12];
  const float* fg_w2     = (const float*)d_in[13];
  const float* fg_b2     = (const float*)d_in[14];
  const float* attn_in_w = (const float*)d_in[15];
  const float* attn_in_b = (const float*)d_in[16];
  const float* attn_out_w= (const float*)d_in[17];
  const float* attn_out_b= (const float*)d_in[18];
  const float* te_conv_w = (const float*)d_in[19];
  const float* te_conv_b = (const float*)d_in[20];
  const float* bn_gamma  = (const float*)d_in[21];
  const float* bn_beta   = (const float*)d_in[22];
  const float* bn_mean   = (const float*)d_in[23];
  const float* bn_var    = (const float*)d_in[24];
  const float* ln1_w = (const float*)d_in[25];
  const float* ln1_b = (const float*)d_in[26];
  const float* ln2_w = (const float*)d_in[27];
  const float* ln2_b = (const float*)d_in[28];
  const float* ln3_w = (const float*)d_in[29];
  const float* ln3_b = (const float*)d_in[30];

  char* ws = (char*)d_ws;
  size_t off = 0;
  auto alloc = [&](size_t bytes) -> char* {
    char* p = ws + off;
    off = (off + bytes + 255) & ~(size_t)255;
    return p;
  };
  _Float16* xh      = (_Float16*)alloc((size_t)BL_ * D_ * 2);
  _Float16* w_inh   = (_Float16*)alloc((size_t)2 * DI_ * D_ * 2);
  _Float16* w_xph   = (_Float16*)alloc((size_t)XDBL_ * DI_ * 2);
  _Float16* w_dtph  = (_Float16*)alloc((size_t)DI_ * DTR_ * 2);
  _Float16* w_outh  = (_Float16*)alloc((size_t)D_ * DI_ * 2);
  _Float16* w_attnh = (_Float16*)alloc((size_t)3 * D_ * D_ * 2);
  float*    uz      = (float*)alloc((size_t)BL_ * 2 * DI_ * 4);
  float*    ucf     = (float*)alloc((size_t)BL_ * DI_ * 4);
  _Float16* uch     = (_Float16*)alloc((size_t)BL_ * DI_ * 2);
  float*    xdbl    = (float*)alloc((size_t)BL_ * XDBL_ * 4);
  _Float16* dth     = (_Float16*)alloc((size_t)BL_ * DTR_ * 2);
  float*    delta   = (float*)alloc((size_t)BL_ * DI_ * 4);
  _Float16* ygh     = (_Float16*)alloc((size_t)BL_ * DI_ * 2);
  float*    x1      = (float*)alloc((size_t)BL_ * D_ * 4);
  float*    ghid    = (float*)alloc((size_t)B_ * 2 * D_ * 4);
  float*    gbuf    = (float*)alloc((size_t)B_ * D_ * 4);
  float*    x2      = (float*)alloc((size_t)BL_ * D_ * 4);
  _Float16* x2h     = (_Float16*)alloc((size_t)BL_ * D_ * 2);
  float*    kbuf    = (float*)alloc((size_t)BL_ * D_ * 4);
  float*    vbuf    = (float*)alloc((size_t)BL_ * D_ * 4);
  float*    qbuf    = (float*)alloc((size_t)B_ * D_ * 4);
  float*    obuf    = (float*)alloc((size_t)B_ * D_ * 4);
  float*    attno   = (float*)alloc((size_t)B_ * D_ * 4);
  float*    x3      = (float*)alloc((size_t)BL_ * D_ * 4);

  auto cvt = [&](const float* src, _Float16* dst, int n) {
    cvt_f32_f16<<<(n + 255) / 256, 256, 0, stream>>>(src, dst, n);
  };
  cvt(x,          xh,      BL_ * D_);
  cvt(in_proj_w,  w_inh,   2 * DI_ * D_);
  cvt(x_proj_w,   w_xph,   XDBL_ * DI_);
  cvt(dt_proj_w,  w_dtph,  DI_ * DTR_);
  cvt(out_proj_w, w_outh,  D_ * DI_);
  cvt(attn_in_w,  w_attnh, 3 * D_ * D_);

  auto gemm = [&](const _Float16* A, const _Float16* W, const float* bias, float* C,
                  int M, int N, int K) {
    dim3 grid(N / 64, M / 64);
    wmma_gemm_nt<<<grid, 128, 0, stream>>>(A, W, bias, C, M, N, K);
  };

  // Mamba path
  gemm(xh, w_inh, nullptr, uz, BL_, 2 * DI_, D_);                               // in_proj
  conv_silu_kernel<<<(BL_ * DI_ + 255) / 256, 256, 0, stream>>>(uz, conv_w, conv_b, ucf, uch);
  gemm(uch, w_xph, nullptr, xdbl, BL_, XDBL_, DI_);                             // x_proj
  slice_dt_kernel<<<(BL_ * DTR_ + 255) / 256, 256, 0, stream>>>(xdbl, dth);
  gemm(dth, w_dtph, nullptr, delta, BL_, DI_, DTR_);                            // dt_proj
  softplus_kernel<<<(BL_ * DI_ + 255) / 256, 256, 0, stream>>>(delta, dt_proj_b);
  scan_kernel<<<(B_ * DI_ * DS_ + 255) / 256, 256, 0, stream>>>(ucf, delta, xdbl, A_log, ssm_D, uz, ygh);
  gemm(ygh, w_outh, nullptr, x1, BL_, D_, DI_);                                 // out_proj

  // freq gate MLP (tiny)
  small_linear_kernel<<<(B_ * 2 * D_ + 255) / 256, 256, 0, stream>>>(freq, fg_w1, fg_b1, ghid, B_, 2 * D_, D_, 1);
  small_linear_kernel<<<(B_ * D_ + 255) / 256, 256, 0, stream>>>(ghid, fg_w2, fg_b2, gbuf, B_, D_, 2 * D_, 2);

  // LN1 * (1+g) + residual -> LN2
  ln1_gate_ln2_kernel<<<BL_, 128, 0, stream>>>(x1, x, gbuf, ln1_w, ln1_b, ln2_w, ln2_b, x2, x2h);

  // cross-attention (q_len = 1)
  small_linear_kernel<<<(B_ * D_ + 255) / 256, 256, 0, stream>>>(freq, attn_in_w, attn_in_b, qbuf, B_, D_, D_, 0);
  gemm(x2h, w_attnh + (size_t)D_ * D_,     attn_in_b + D_,     kbuf, BL_, D_, D_); // K proj
  gemm(x2h, w_attnh + (size_t)2 * D_ * D_, attn_in_b + 2 * D_, vbuf, BL_, D_, D_); // V proj
  attn_kernel<<<B_ * H_, 256, 0, stream>>>(qbuf, kbuf, vbuf, obuf);
  small_linear_kernel<<<(B_ * D_ + 255) / 256, 256, 0, stream>>>(obuf, attn_out_w, attn_out_b, attno, B_, D_, D_, 0);

  // LN3, then grouped temporal conv + BN + GELU residual -> out
  ln3_kernel<<<BL_, 128, 0, stream>>>(x2, attno, ln3_w, ln3_b, x3);
  teconv_kernel<<<(BL_ * D_ + 255) / 256, 256, 0, stream>>>(x3, te_conv_w, te_conv_b, bn_gamma,
                                                            bn_beta, bn_mean, bn_var, (float*)d_out);
}
